// PaddedAtomQueryFieldNet_55903294325330
// MI455X (gfx1250) — matrix operand order repaired and checked by reading, not compiled, not run on hardware
//
#include <hip/hip_runtime.h>

typedef _Float16     v16h __attribute__((ext_vector_type(16)));
typedef _Float16     h2   __attribute__((ext_vector_type(2)));
typedef float        v8f  __attribute__((ext_vector_type(8)));
typedef unsigned int v8u  __attribute__((ext_vector_type(8)));

#define A_ATOMS    40
#define CUTOFF_F   8.0f
#define EPS_F      1e-12f
#define PACK_HALFS (96 * 512)       // 96 A-fragments x 512 f16
#define SW_OFFSET  (96 * 1024)      // bytes: packed weights occupy 96 KB of d_ws

// cvt_pk two floats to f16 pair, then packed relu (v_cvt_pk_f16_f32 + v_pk_max_num_f16)
static __device__ inline unsigned pk2relu(float a, float b) {
  h2 v;
  v[0] = (_Float16)a;
  v[1] = (_Float16)b;
  h2 z = {};
  h2 m = __builtin_elementwise_max(v, z);
  return __builtin_bit_cast(unsigned, m);
}

// lane L <-> lane L^16 exchange: ds_swizzle_b32 SWAPX16 (xor=0x10, and=0x1f)
static __device__ inline unsigned swap16(unsigned x) {
  return (unsigned)__builtin_amdgcn_ds_swizzle((int)x, 0x401F);
}
static __device__ inline float swap16f(float x) {
  return __builtin_bit_cast(float,
      (unsigned)__builtin_amdgcn_ds_swizzle((int)__builtin_bit_cast(unsigned, x), 0x401F));
}

// ---------------------------------------------------------------------------
// Kernel 0: pack f32 weights (transposed) into f16 A-fragment layout.
// A-frag (M=16 x K=32): lane L -> m = L%16, h = L/16;
//   element i=2v+e -> k = 16h + 2v + e (contiguous)
// value = W[k][m]  (i.e. Wt[m][k]); layer-1 additionally embeds bias at k==34.
// Fragment order per matrix: frag = kc*mtiles + mt.
// Bases: W1 frags [0,16), W2 [16,48), W3 [48,80), Ws1 [80,96).
// ---------------------------------------------------------------------------
__global__ __launch_bounds__(256) void pack_weights_kernel(
    const float* __restrict__ w1, const float* __restrict__ b1,
    const float* __restrict__ w2, const float* __restrict__ w3,
    const float* __restrict__ ws1, _Float16* __restrict__ pack)
{
  int tid = blockIdx.x * blockDim.x + threadIdx.x;
  if (tid >= PACK_HALFS) return;
  int fragIdx = tid >> 9;
  int within  = tid & 511;
  int lane = within >> 4;
  int idx  = within & 15;
  int v = idx >> 1, e = idx & 1;
  int ml = lane & 15, hh = lane >> 4;
  int klocal = 16 * hh + 2 * v + e;

  float val;
  if (fragIdx < 16) {                       // layer 1: (34+bias) x 128
    int f = fragIdx, kc = f >> 3, mt = f & 7;
    int k = kc * 32 + klocal, m = mt * 16 + ml;
    val = (k < 34) ? w1[k * 128 + m] : ((k == 34) ? b1[m] : 0.0f);
  } else if (fragIdx < 48) {                // layer 2: 128 x 128
    int f = fragIdx - 16, kc = f >> 3, mt = f & 7;
    int k = kc * 32 + klocal, m = mt * 16 + ml;
    val = w2[k * 128 + m];
  } else if (fragIdx < 80) {                // layer 3: 128 x 128
    int f = fragIdx - 48, kc = f >> 3, mt = f & 7;
    int k = kc * 32 + klocal, m = mt * 16 + ml;
    val = w3[k * 128 + m];
  } else {                                  // scalar head: 128 x 64
    int f = fragIdx - 80, kc = f >> 2, mt = f & 3;
    int k = kc * 32 + klocal, m = mt * 16 + ml;
    val = ws1[k * 64 + m];
  }
  pack[tid] = (_Float16)val;
}

// ---------------------------------------------------------------------------
// Kernel 1: per-pair MLP via WMMA, transposed form D = Wt (A) x Xt (B).
// One wave = TWO 16-pair tiles: each weight A-fragment is loaded once and
// feeds two WMMAs. Weight fragments are explicitly double-buffered so each
// s_wait_loadcnt hides behind a full tile of WMMA work. Each tile's D is
// packed (relu+f16) right after accumulation (4 dwords/tile live). Inter-
// layer exchange is one ds_swizzle SWAPX16 per dword.
// ---------------------------------------------------------------------------
__global__ __launch_bounds__(256) void pair_mlp_kernel(
    const float* __restrict__ atom_pos, const float* __restrict__ atom_feat,
    const float* __restrict__ query_pos, const float* __restrict__ atom_mask,
    const float* __restrict__ gammap, const float* __restrict__ centers,
    const float* __restrict__ b2, const float* __restrict__ b3,
    const float* __restrict__ bs1, const float* __restrict__ ws2,
    const float* __restrict__ bs2,
    const _Float16* __restrict__ pack, float* __restrict__ scalar_w, int P)
{
  const int lane = threadIdx.x & 31;
  const int wv   = threadIdx.x >> 5;
  const int r    = lane & 15;       // pair column within tile
  const int h    = lane >> 4;       // half-wave
  const int pb   = (blockIdx.x * 8 + wv) * 32;   // 2 tiles of 16 pairs
  if (pb >= P) return;              // wave-uniform

  const float gamma = gammap[0];

  int   pp[2];
  bool  valid[2];
  int   aidx[2];
  float dd[2];
  v16h  xb[2][2];

#pragma unroll
  for (int u = 0; u < 2; ++u) {
    int p = pb + u * 16 + r;
    valid[u] = (p < P);
    if (p >= P) p = P - 1;          // clamp, no divergence
    pp[u] = p;
    const int q = p / A_ATOMS;
    const int a = p - q * A_ATOMS;
    aidx[u] = a;

    const float qx = query_pos[q * 3 + 0], qy = query_pos[q * 3 + 1], qz = query_pos[q * 3 + 2];
    const float ax = atom_pos[a * 3 + 0],  ay = atom_pos[a * 3 + 1],  az = atom_pos[a * 3 + 2];
    const float rx = qx - ax + EPS_F, ry = qy - ay + EPS_F, rz = qz - az + EPS_F;
    const float dist = sqrtf(rx * rx + ry * ry + rz * rz);
    dd[u] = dist;
    const float f0 = atom_feat[a * 2 + 0], f1 = atom_feat[a * 2 + 1];

    // ---- layer-1 B fragments (Xt): column = own pair, k = feature ----
    // B layout: element i=2v+e -> k = 16h + 2v + e (within 32-k chunk)
    // chunk 0: k = 16h + 0..15 (< 34); features only at k=0,1 (h==0, v==0)
#pragma unroll
    for (int v = 0; v < 8; ++v) {
#pragma unroll
      for (int e = 0; e < 2; ++e) {
        int k = 16 * h + 2 * v + e;
        int ci = k - 2;
        ci = ci < 0 ? 0 : ci;               // safe index when k<2
        const float d = dist - centers[ci];
        float x = __expf(-gamma * d * d);
        if (v == 0) {
          const float fe = (e == 0) ? f0 : f1;
          x = (h == 0) ? fe : x;
        }
        xb[u][0][v * 2 + e] = (_Float16)x;
      }
    }
    // chunk 1: nonzero only for h==0: k=32,33 (RBF), k=34 (bias row = 1.0)
    {
      v16h t = {};
      const float d30 = dist - centers[30];
      const float d31 = dist - centers[31];
      const float r30 = __expf(-gamma * d30 * d30);
      const float r31 = __expf(-gamma * d31 * d31);
      t[0] = (h == 0) ? (_Float16)r30 : (_Float16)0.0f;
      t[1] = (h == 0) ? (_Float16)r31 : (_Float16)0.0f;
      t[2] = (h == 0) ? (_Float16)1.0f : (_Float16)0.0f;
      xb[u][1] = t;
    }
  }

  unsigned pkbuf[2][8][4];   // packed relu'd f16 activations (post-layer)
  v16h     bf[2][4];         // B fragments for the current layer

  // pkbuf -> B fragments: one ds_swizzle SWAPX16 + cndmask per dword.
  // Own half provides 8 feature-rows of tile (2c+h); partner provides the rest.
  auto make_bfrags = [&](int u) {
#pragma unroll
    for (int c = 0; c < 4; ++c) {
      v8u bu;
#pragma unroll
      for (int j2 = 0; j2 < 4; ++j2) {
        const unsigned y_hi = swap16(pkbuf[u][2 * c][j2]);      // o=8..15 for h==0
        const unsigned y_lo = swap16(pkbuf[u][2 * c + 1][j2]);  // o=0..7  for h==1
        bu[j2]     = h ? y_lo : pkbuf[u][2 * c][j2];
        bu[j2 + 4] = h ? pkbuf[u][2 * c + 1][j2] : y_hi;
      }
      bf[u][c] = __builtin_bit_cast(v16h, bu);
    }
  };

  // ---- layer 1: Wt1 (128 x 64, bias folded at k=34) x Xt (64 x 16) x2 ----
  {
    v16h af[2][2];
#pragma unroll
    for (int kc = 0; kc < 2; ++kc)
      af[0][kc] = *(const v16h*)(pack + (kc * 8 + 0) * 512 + lane * 16);
#pragma unroll
    for (int mt = 0; mt < 8; ++mt) {
      const int cur = mt & 1, nxt = cur ^ 1;
      if (mt < 7) {
#pragma unroll
        for (int kc = 0; kc < 2; ++kc)
          af[nxt][kc] = *(const v16h*)(pack + (kc * 8 + mt + 1) * 512 + lane * 16);
      }
      v8f c0 = {}, c1 = {};
#pragma unroll
      for (int kc = 0; kc < 2; ++kc) {
        c0 = __builtin_amdgcn_wmma_f32_16x16x32_f16(false, af[cur][kc], false, xb[0][kc],
                                                    (short)0, c0, false, false);
        c1 = __builtin_amdgcn_wmma_f32_16x16x32_f16(false, af[cur][kc], false, xb[1][kc],
                                                    (short)0, c1, false, false);
      }
#pragma unroll
      for (int j2 = 0; j2 < 4; ++j2) {
        pkbuf[0][mt][j2] = pk2relu(c0[2 * j2], c0[2 * j2 + 1]);
        pkbuf[1][mt][j2] = pk2relu(c1[2 * j2], c1[2 * j2 + 1]);
      }
    }
  }

  // ---- layers 2 and 3: Wt (128x128) x H (128x16) x2; bias via C load ----
  auto run_layer = [&](const _Float16* wbase, const float* bias) {
    make_bfrags(0);
    make_bfrags(1);
    v16h af[2][4];
#pragma unroll
    for (int kc = 0; kc < 4; ++kc)
      af[0][kc] = *(const v16h*)(wbase + (kc * 8 + 0) * 512 + lane * 16);
#pragma unroll
    for (int mt = 0; mt < 8; ++mt) {
      const int cur = mt & 1, nxt = cur ^ 1;
      if (mt < 7) {
#pragma unroll
        for (int kc = 0; kc < 4; ++kc)
          af[nxt][kc] = *(const v16h*)(wbase + (kc * 8 + mt + 1) * 512 + lane * 16);
      }
      v8f c0 = *(const v8f*)(bias + mt * 16 + 8 * h);   // rows m = 16mt+8h+j
      v8f c1 = c0;
#pragma unroll
      for (int kc = 0; kc < 4; ++kc) {
        c0 = __builtin_amdgcn_wmma_f32_16x16x32_f16(false, af[cur][kc], false, bf[0][kc],
                                                    (short)0, c0, false, false);
        c1 = __builtin_amdgcn_wmma_f32_16x16x32_f16(false, af[cur][kc], false, bf[1][kc],
                                                    (short)0, c1, false, false);
      }
#pragma unroll
      for (int j2 = 0; j2 < 4; ++j2) {
        pkbuf[0][mt][j2] = pk2relu(c0[2 * j2], c0[2 * j2 + 1]);
        pkbuf[1][mt][j2] = pk2relu(c1[2 * j2], c1[2 * j2 + 1]);
      }
    }
  };
  run_layer(pack + 16 * 512, b2);
  run_layer(pack + 48 * 512, b3);

  // ---- scalar head: Wts1 (64x128) x H (128x16) x2, then dot with ws2 ----
  make_bfrags(0);
  make_bfrags(1);
  float partial[2] = {0.0f, 0.0f};
  {
    const _Float16* hbase = pack + 80 * 512;
    v16h af[2][4];
#pragma unroll
    for (int kc = 0; kc < 4; ++kc)
      af[0][kc] = *(const v16h*)(hbase + (kc * 4 + 0) * 512 + lane * 16);
#pragma unroll
    for (int mt = 0; mt < 4; ++mt) {
      const int cur = mt & 1, nxt = cur ^ 1;
      if (mt < 3) {
#pragma unroll
        for (int kc = 0; kc < 4; ++kc)
          af[nxt][kc] = *(const v16h*)(hbase + (kc * 4 + mt + 1) * 512 + lane * 16);
      }
      v8f c0 = *(const v8f*)(bs1 + mt * 16 + 8 * h);
      v8f c1 = c0;
#pragma unroll
      for (int kc = 0; kc < 4; ++kc) {
        c0 = __builtin_amdgcn_wmma_f32_16x16x32_f16(false, af[cur][kc], false, bf[0][kc],
                                                    (short)0, c0, false, false);
        c1 = __builtin_amdgcn_wmma_f32_16x16x32_f16(false, af[cur][kc], false, bf[1][kc],
                                                    (short)0, c1, false, false);
      }
      const v8f w8 = *(const v8f*)(ws2 + mt * 16 + 8 * h);
#pragma unroll
      for (int j = 0; j < 8; ++j) {
        partial[0] += fmaxf(c0[j], 0.0f) * w8[j];
        partial[1] += fmaxf(c1[j], 0.0f) * w8[j];
      }
    }
  }
  // each half summed 32 of the 64 features of its own pair; combine halves
  const float bias2 = bs2[0];
#pragma unroll
  for (int u = 0; u < 2; ++u) {
    const float total = partial[u] + swap16f(partial[u]);
    if (h == 0 && valid[u]) {
      const float mk = atom_mask[aidx[u]] * (dd[u] <= CUTOFF_F ? 1.0f : 0.0f);
      scalar_w[pp[u]] = (total + bias2) * mk;
    }
  }
}

// ---------------------------------------------------------------------------
// Kernel 2: per-query weighted direction sum + final 3->64->64->3 MLP
// ---------------------------------------------------------------------------
__global__ __launch_bounds__(256) void finalize_kernel(
    const float* __restrict__ atom_pos, const float* __restrict__ query_pos,
    const float* __restrict__ wf1, const float* __restrict__ bf1,
    const float* __restrict__ wf2, const float* __restrict__ bf2,
    const float* __restrict__ wf3, const float* __restrict__ bf3,
    const float* __restrict__ scalar_w, float* __restrict__ out, int Nq)
{
  int q = blockIdx.x * blockDim.x + threadIdx.x;
  if (q >= Nq) return;
  const float qx = query_pos[q * 3 + 0], qy = query_pos[q * 3 + 1], qz = query_pos[q * 3 + 2];
  float wx = 0.0f, wy = 0.0f, wz = 0.0f;
  for (int a = 0; a < A_ATOMS; ++a) {
    const float rx = qx - atom_pos[a * 3 + 0];
    const float ry = qy - atom_pos[a * 3 + 1];
    const float rz = qz - atom_pos[a * 3 + 2];
    const float ex = rx + EPS_F, ey = ry + EPS_F, ez = rz + EPS_F;
    const float dist = sqrtf(ex * ex + ey * ey + ez * ez);
    const float inv = 1.0f / (dist + EPS_F);
    const float s = scalar_w[q * A_ATOMS + a];
    wx += s * rx * inv; wy += s * ry * inv; wz += s * rz * inv;
  }
  float o1[64];
#pragma unroll
  for (int j = 0; j < 64; ++j) {
    float t = bf1[j] + wx * wf1[j] + wy * wf1[64 + j] + wz * wf1[128 + j];
    o1[j] = fmaxf(t, 0.0f);
  }
  float a0 = bf3[0], a1 = bf3[1], a2 = bf3[2];
  for (int j = 0; j < 64; ++j) {        // dynamic j; o2 consumed on the fly
    float t = bf2[j];
#pragma unroll
    for (int i = 0; i < 64; ++i) t += o1[i] * wf2[i * 64 + j];
    t = fmaxf(t, 0.0f);
    a0 += t * wf3[j * 3 + 0];
    a1 += t * wf3[j * 3 + 1];
    a2 += t * wf3[j * 3 + 2];
  }
  out[q * 3 + 0] = a0;
  out[q * 3 + 1] = a1;
  out[q * 3 + 2] = a2;
}

// ---------------------------------------------------------------------------
extern "C" void kernel_launch(void* const* d_in, const int* in_sizes, int n_in,
                              void* d_out, int out_size, void* d_ws, size_t ws_size,
                              hipStream_t stream)
{
  const float* atom_pos  = (const float*)d_in[0];
  const float* atom_feat = (const float*)d_in[1];
  const float* query_pos = (const float*)d_in[2];
  const float* atom_mask = (const float*)d_in[3];
  const float* gamma     = (const float*)d_in[4];
  const float* centers   = (const float*)d_in[5];
  const float* w1  = (const float*)d_in[6];  const float* b1  = (const float*)d_in[7];
  const float* w2  = (const float*)d_in[8];  const float* b2  = (const float*)d_in[9];
  const float* w3  = (const float*)d_in[10]; const float* b3  = (const float*)d_in[11];
  const float* ws1 = (const float*)d_in[12]; const float* bs1 = (const float*)d_in[13];
  const float* ws2 = (const float*)d_in[14]; const float* bs2 = (const float*)d_in[15];
  const float* wf1 = (const float*)d_in[16]; const float* bf1 = (const float*)d_in[17];
  const float* wf2 = (const float*)d_in[18]; const float* bf2 = (const float*)d_in[19];
  const float* wf3 = (const float*)d_in[20]; const float* bf3 = (const float*)d_in[21];

  const int Nq = in_sizes[2] / 3;
  const int P  = Nq * A_ATOMS;

  _Float16* pack    = (_Float16*)d_ws;
  float*    scalarw = (float*)((char*)d_ws + SW_OFFSET);
  float*    outf    = (float*)d_out;

  pack_weights_kernel<<<(PACK_HALFS + 255) / 256, 256, 0, stream>>>(w1, b1, w2, w3, ws1, pack);

  // 8 waves per block, 32 pairs per wave
  const int blocks = (P + 255) / 256;
  pair_mlp_kernel<<<blocks, 256, 0, stream>>>(atom_pos, atom_feat, query_pos, atom_mask,
                                              gamma, centers, b2, b3, bs1, ws2, bs2,
                                              pack, scalarw, P);

  finalize_kernel<<<(Nq + 255) / 256, 256, 0, stream>>>(atom_pos, query_pos,
                                                        wf1, bf1, wf2, bf2, wf3, bf3,
                                                        scalarw, outf, Nq);
}
